// Aggregationfeature_22995254903075
// MI455X (gfx1250) — matrix operand order repaired and checked by reading, not compiled, not run on hardware
//
#include <hip/hip_runtime.h>

#define Nn 100000
#define Ee 400000
#define Rr 4
#define Gg 64
#define INf 64
#define Hh 128
#define Cc 2

typedef __attribute__((ext_vector_type(16))) _Float16 v16h;
typedef __attribute__((ext_vector_type(8)))  float    v8f;

// ---------------- utility kernels ----------------

__global__ void k_zero(float* __restrict__ p, int n) {
  int i = blockIdx.x * blockDim.x + threadIdx.x;
  if (i < n) p[i] = 0.0f;
}

__global__ void k_degree(const int* __restrict__ src, const int* __restrict__ dst,
                         float* __restrict__ degO, float* __restrict__ degI, int E) {
  int e = blockIdx.x * blockDim.x + threadIdx.x;
  if (e < E) {
    atomicAdd(&degO[src[e]], 1.0f);
    atomicAdd(&degI[dst[e]], 1.0f);
  }
}

__global__ void k_rsqrt(float* __restrict__ p, int n) {
  int i = blockIdx.x * blockDim.x + threadIdx.x;
  if (i < n) p[i] = rsqrtf(fmaxf(p[i], 1.0f));
}

// ---------------- WMMA GEMM: C[M,128] = (rowscale .* A[M,K]) @ B[K,128] (+bias)(+relu) ----------------

__global__ void __launch_bounds__(256) k_gemm_wmma(
    const float* __restrict__ A, const float* __restrict__ Bw,
    const float* __restrict__ bias, const float* __restrict__ rowscale,
    float* __restrict__ Cout, int M, int K, int doRelu)
{
  // K staged in 64-wide chunks; LDS padded to 68 halfs/row to dodge bank conflicts
  __shared__ _Float16 As[128 * 68];   // A tile: 128 rows x 64 k   (row-major)
  __shared__ _Float16 Bs[128 * 68];   // B tile: 128 cols x 64 k   (transposed)
  const int LD = 68;

  const int tid  = threadIdx.x;
  const int lane = tid & 31;
  const int w    = tid >> 5;          // wave 0..7 -> 16-row slab
  const int half = lane >> 4;         // 0/1
  const int lr   = lane & 15;         // 0..15
  const int blockM = blockIdx.x * 128;

  v8f acc[8];
#pragma unroll
  for (int i = 0; i < 8; ++i) {
    v8f z = {0.f,0.f,0.f,0.f,0.f,0.f,0.f,0.f};
    acc[i] = z;
  }

  for (int k0 = 0; k0 < K; k0 += 64) {
    // stage A chunk: 128 x 64, f32 -> f16, optional per-row scale
    for (int i = tid; i < 128 * 64; i += 256) {
      int m = i >> 6, k = i & 63;
      int gr = blockM + m;
      float v = 0.0f;
      if (gr < M) {
        v = A[(size_t)gr * K + (k0 + k)];
        if (rowscale) v *= rowscale[gr];
      }
      As[m * LD + k] = (_Float16)v;
    }
    // stage B chunk: 64 x 128 row-major in global -> Bs[n][k]
    for (int i = tid; i < 64 * 128; i += 256) {
      int k = i >> 7, n = i & 127;
      Bs[n * LD + k] = (_Float16)Bw[(size_t)(k0 + k) * 128 + n];
    }
    __syncthreads();

#pragma unroll
    for (int kk = 0; kk < 64; kk += 32) {
      // A fragment (16x32): lanes 0-15: K {kk+0..7, kk+16..23}; lanes 16-31: +8
      v16h af;
      const int m = w * 16 + lr;
#pragma unroll
      for (int p = 0; p < 4; ++p) {
        int kl = kk + half * 8 + 2 * p;
        af[2 * p]     = As[m * LD + kl];
        af[2 * p + 1] = As[m * LD + kl + 1];
      }
#pragma unroll
      for (int p = 0; p < 4; ++p) {
        int kl = kk + 16 + half * 8 + 2 * p;
        af[8 + 2 * p]     = As[m * LD + kl];
        af[8 + 2 * p + 1] = As[m * LD + kl + 1];
      }
#pragma unroll
      for (int nt = 0; nt < 8; ++nt) {
        // B fragment (32x16): lane = column, lanes 0-15: K {kk+0..7, kk+16..23}; lanes 16-31: +8
        v16h bf;
        const int n = nt * 16 + lr;
#pragma unroll
        for (int p = 0; p < 8; ++p) {
          int kl = kk + (p >> 2) * 16 + half * 8 + 2 * (p & 3);
          bf[2 * p]     = Bs[n * LD + kl];
          bf[2 * p + 1] = Bs[n * LD + kl + 1];
        }
        acc[nt] = __builtin_amdgcn_wmma_f32_16x16x32_f16(
            false, af, false, bf, (short)0, acc[nt], false, false);
      }
    }
    __syncthreads();
  }

  // epilogue: C layout — VGPR v holds M = half*8 + v, lane lr = column within tile
#pragma unroll
  for (int nt = 0; nt < 8; ++nt) {
    const int col = nt * 16 + lr;
    const float bv = bias ? bias[col] : 0.0f;
#pragma unroll
    for (int v = 0; v < 8; ++v) {
      int row = blockM + w * 16 + half * 8 + v;
      if (row < M) {
        float val = acc[nt][v] + bv;
        if (doRelu) val = fmaxf(val, 0.0f);
        Cout[(size_t)row * 128 + col] = val;
      }
    }
  }
}

// ---------------- edge scatter: out[dst] += rinI[dst] * Z[src] ----------------

__global__ void k_scatter(const int* __restrict__ src, const int* __restrict__ dst,
                          const float* __restrict__ Z, const float* __restrict__ rinI,
                          float* __restrict__ out, int E)
{
  long gid = (long)blockIdx.x * blockDim.x + threadIdx.x;
  if (gid >= (long)E * Hh) return;
  int e = (int)(gid >> 7);
  int k = (int)(gid & 127);
  int s = src[e], d = dst[e];
  atomicAdd(&out[(size_t)d * Hh + k], rinI[d] * Z[(size_t)s * Hh + k]);
}

// ---------------- sum-of-relation-biases + optional relu ----------------

__global__ void k_bias_act(const float* __restrict__ in, const float* __restrict__ bR,
                           float* __restrict__ out, int n, int doRelu)
{
  int i = blockIdx.x * blockDim.x + threadIdx.x;
  if (i < n) {
    int k = i & 127;
    float bs = bR[k] + bR[Hh + k] + bR[2 * Hh + k] + bR[3 * Hh + k];
    float v = in[i] + bs;
    out[i] = doRelu ? fmaxf(v, 0.0f) : v;
  }
}

// ---------------- per-graph mean pooling ----------------

__global__ void k_count(const int* __restrict__ gids, float* __restrict__ counts, int n) {
  int i = blockIdx.x * blockDim.x + threadIdx.x;
  if (i < n) atomicAdd(&counts[gids[i]], 1.0f);
}

__global__ void k_pool(const float* __restrict__ h, const int* __restrict__ gids,
                       float* __restrict__ pooled, int n)
{
  long gid = (long)blockIdx.x * blockDim.x + threadIdx.x;
  if (gid >= (long)n * Hh) return;
  int node = (int)(gid >> 7);
  int k = (int)(gid & 127);
  atomicAdd(&pooled[(size_t)gids[node] * Hh + k], h[gid]);
}

__global__ void k_pooldiv(float* __restrict__ pooled, const float* __restrict__ counts, int n) {
  int i = blockIdx.x * blockDim.x + threadIdx.x;
  if (i < n) pooled[i] /= fmaxf(counts[i >> 7], 1.0f);
}

// ---------------- MLP head: [64,128]@[128,128] relu x2 -> [128,2] ----------------

__global__ void __launch_bounds__(128) k_head(
    const float* __restrict__ pooled,
    const float* __restrict__ Wm1, const float* __restrict__ bm1,
    const float* __restrict__ Wm2, const float* __restrict__ bm2,
    const float* __restrict__ Wm3, const float* __restrict__ bm3,
    float* __restrict__ out)
{
  __shared__ float r0[Hh], z1[Hh], z2[Hh];
  int g = blockIdx.x, t = threadIdx.x;
  r0[t] = pooled[(size_t)g * Hh + t];
  __syncthreads();
  float s = bm1[t];
  for (int k = 0; k < Hh; ++k) s += r0[k] * Wm1[k * Hh + t];
  z1[t] = fmaxf(s, 0.0f);
  __syncthreads();
  s = bm2[t];
  for (int k = 0; k < Hh; ++k) s += z1[k] * Wm2[k * Hh + t];
  z2[t] = fmaxf(s, 0.0f);
  __syncthreads();
  if (t < Cc) {
    s = bm3[t];
    for (int k = 0; k < Hh; ++k) s += z2[k] * Wm3[k * Cc + t];
    out[(size_t)g * Cc + t] = s;
  }
}

// ---------------- launch ----------------

extern "C" void kernel_launch(void* const* d_in, const int* in_sizes, int n_in,
                              void* d_out, int out_size, void* d_ws, size_t ws_size,
                              hipStream_t stream)
{
  (void)in_sizes; (void)n_in; (void)out_size; (void)ws_size;

  const float* x    = (const float*)d_in[0];
  const int*   src  = (const int*)d_in[1];   // [R,E]
  const int*   dst  = (const int*)d_in[2];   // [R,E]
  const int*   gids = (const int*)d_in[3];   // [N]
  const float* W_in = (const float*)d_in[4];
  const float* b_in = (const float*)d_in[5];
  const float* W1   = (const float*)d_in[6]; // [R,H,H]
  const float* b1   = (const float*)d_in[7]; // [R,H]
  const float* W2   = (const float*)d_in[8];
  const float* b2   = (const float*)d_in[9];
  const float* Wm1  = (const float*)d_in[10];
  const float* bm1  = (const float*)d_in[11];
  const float* Wm2  = (const float*)d_in[12];
  const float* bm2  = (const float*)d_in[13];
  const float* Wm3  = (const float*)d_in[14];
  const float* bm3  = (const float*)d_in[15];
  float* out = (float*)d_out;

  // workspace layout (floats)
  float* hA     = (float*)d_ws;                 // [N,H] features (layer input / final)
  float* hB     = hA + (size_t)Nn * Hh;         // [N,H] scatter accumulator
  float* Z      = hB + (size_t)Nn * Hh;         // [N,H] per-relation GEMM result
  float* rinO   = Z  + (size_t)Nn * Hh;         // [R,N] rsqrt(out-degree)
  float* rinI   = rinO + (size_t)Rr * Nn;       // [R,N] rsqrt(in-degree)
  float* pooled = rinI + (size_t)Rr * Nn;       // [G,H]
  float* counts = pooled + Gg * Hh;             // [G]

  const dim3 T(256);
  auto B = [](long n) { return dim3((unsigned)((n + 255) / 256)); };
  const int gemmGrid = (Nn + 127) / 128;

  // degree normalizers (shared by both layers)
  k_zero<<<B(2L * Rr * Nn), T, 0, stream>>>(rinO, 2 * Rr * Nn);
  for (int r = 0; r < Rr; ++r)
    k_degree<<<B(Ee), T, 0, stream>>>(src + (size_t)r * Ee, dst + (size_t)r * Ee,
                                      rinO + (size_t)r * Nn, rinI + (size_t)r * Nn, Ee);
  k_rsqrt<<<B(2L * Rr * Nn), T, 0, stream>>>(rinO, 2 * Rr * Nn);

  // h = relu(x @ W_in + b_in)
  k_gemm_wmma<<<dim3(gemmGrid), T, 0, stream>>>(x, W_in, b_in, (const float*)nullptr,
                                                hA, Nn, INf, 1);

  // layer 1: hB = sum_r D_in^-1/2 A_r^T (D_out^-1/2 h) W1_r ; hA = relu(hB + sum_r b1_r)
  k_zero<<<B((long)Nn * Hh), T, 0, stream>>>(hB, Nn * Hh);
  for (int r = 0; r < Rr; ++r) {
    k_gemm_wmma<<<dim3(gemmGrid), T, 0, stream>>>(hA, W1 + (size_t)r * Hh * Hh,
                                                  (const float*)nullptr,
                                                  rinO + (size_t)r * Nn, Z, Nn, Hh, 0);
    k_scatter<<<B((long)Ee * Hh), T, 0, stream>>>(src + (size_t)r * Ee, dst + (size_t)r * Ee,
                                                  Z, rinI + (size_t)r * Nn, hB, Ee);
  }
  k_bias_act<<<B((long)Nn * Hh), T, 0, stream>>>(hB, b1, hA, Nn * Hh, 1);

  // layer 2 (no relu)
  k_zero<<<B((long)Nn * Hh), T, 0, stream>>>(hB, Nn * Hh);
  for (int r = 0; r < Rr; ++r) {
    k_gemm_wmma<<<dim3(gemmGrid), T, 0, stream>>>(hA, W2 + (size_t)r * Hh * Hh,
                                                  (const float*)nullptr,
                                                  rinO + (size_t)r * Nn, Z, Nn, Hh, 0);
    k_scatter<<<B((long)Ee * Hh), T, 0, stream>>>(src + (size_t)r * Ee, dst + (size_t)r * Ee,
                                                  Z, rinI + (size_t)r * Nn, hB, Ee);
  }
  k_bias_act<<<B((long)Nn * Hh), T, 0, stream>>>(hB, b2, hA, Nn * Hh, 0);

  // per-graph mean pooling
  k_zero<<<B(Gg * Hh + Gg), T, 0, stream>>>(pooled, Gg * Hh + Gg);
  k_count<<<B(Nn), T, 0, stream>>>(gids, counts, Nn);
  k_pool<<<B((long)Nn * Hh), T, 0, stream>>>(hA, gids, pooled, Nn);
  k_pooldiv<<<B(Gg * Hh), T, 0, stream>>>(pooled, counts, Gg * Hh);

  // MLP head -> [G, C]
  k_head<<<dim3(Gg), dim3(128), 0, stream>>>(pooled, Wm1, bm1, Wm2, bm2, Wm3, bm3, out);
}
